// GCViT_28089086116135
// MI455X (gfx1250) — matrix operand order repaired and checked by reading, not compiled, not run on hardware
//
#include <hip/hip_runtime.h>
#include <hip/hip_bf16.h>
#include <stdint.h>

// ---------------------------------------------------------------------------
// CDNA5 (gfx1250) wave32 WMMA types / helpers
// ---------------------------------------------------------------------------
typedef __attribute__((ext_vector_type(16))) _Float16 v16h;
typedef __attribute__((ext_vector_type(8)))  float    v8f;
typedef __attribute__((ext_vector_type(4)))  int      i32x4_t;

#define WMMA_F16(a, b, c) \
  __builtin_amdgcn_wmma_f32_16x16x32_f16(false, (a), false, (b), (short)0, (c), false, false)

#if defined(__has_builtin)
#if __has_builtin(__builtin_amdgcn_global_load_async_to_lds_b128)
#define HAS_ASYNC_LDS 1
#endif
#endif

// 16-byte global -> LDS copy; async (ASYNCcnt-tracked, no VGPR data path)
// when the toolchain exposes the gfx1250 builtin.
__device__ __forceinline__ void copy16_g2l(const _Float16* g, _Float16* l) {
#ifdef HAS_ASYNC_LDS
  __builtin_amdgcn_global_load_async_to_lds_b128(
      (__attribute__((address_space(1))) i32x4_t*)g,
      (__attribute__((address_space(3))) i32x4_t*)l, 0, 0);
#else
  *(uint4*)l = *(const uint4*)g;
#endif
}

__device__ __forceinline__ void wait_async_copies() {
#ifdef HAS_ASYNC_LDS
#if __has_builtin(__builtin_amdgcn_s_wait_asynccnt)
  __builtin_amdgcn_s_wait_asynccnt(0);
#else
  asm volatile("s_wait_asynccnt 0x0" ::: "memory");
#endif
#endif
}

// A-matrix fragment (16x32 f16), source row-major [16][K] stride ld (halfs).
// ISA 7.12.2: lanes 0-15 row m=lane, K={0..7,16..23}; lanes 16-31 K={8..15,24..31}.
__device__ __forceinline__ v16h frag_a(const _Float16* p, int ld, int lane) {
  int m  = lane & 15;
  int k0 = (lane >> 4) * 8;
  const _Float16* r = p + m * ld;
  v16h a;
#pragma unroll
  for (int e = 0; e < 8; ++e) a[e] = r[k0 + e];
#pragma unroll
  for (int e = 0; e < 8; ++e) a[8 + e] = r[16 + k0 + e];
  return a;
}

// B-matrix fragment (32x16 f16) from a TRANSPOSED source: src[n][k], stride ld.
// Lanes 0-15: K={0..7,16..23}, lanes 16-31: K={8..15,24..31} (matches frag_a).
__device__ __forceinline__ v16h frag_bT(const _Float16* p, int ld, int lane) {
  int n  = lane & 15;
  int k0 = (lane >> 4) * 8;
  const _Float16* r = p + n * ld;
  v16h b;
#pragma unroll
  for (int e = 0; e < 8; ++e) b[e] = r[k0 + e];
#pragma unroll
  for (int e = 0; e < 8; ++e) b[8 + e] = r[16 + k0 + e];
  return b;
}

// ---------------------------------------------------------------------------
// f32 -> f16 conversion (weights)
// ---------------------------------------------------------------------------
__global__ __launch_bounds__(256) void cvt_f16(const float* __restrict__ in,
                                               _Float16* __restrict__ out, int n) {
  int i = blockIdx.x * 256 + threadIdx.x;
  if (i < n) out[i] = (_Float16)in[i];
}

// ---------------------------------------------------------------------------
// LayerNorm (C=192), one wave per token; optional window-partition remap.
// ---------------------------------------------------------------------------
__global__ __launch_bounds__(256) void ln_kernel(const float* __restrict__ x,
                                                 const float* __restrict__ g,
                                                 const float* __restrict__ bta,
                                                 _Float16* __restrict__ out,
                                                 int windowed) {
  int tok  = blockIdx.x * 8 + (threadIdx.x >> 5);
  int lane = threadIdx.x & 31;
  const float* px = x + (size_t)tok * 192;
  float vals[6];
  float s = 0.f, s2 = 0.f;
#pragma unroll
  for (int j = 0; j < 6; ++j) {
    float v = px[lane + 32 * j];
    vals[j] = v; s += v; s2 += v * v;
  }
#pragma unroll
  for (int off = 16; off >= 1; off >>= 1) {
    s  += __shfl_xor(s, off, 32);
    s2 += __shfl_xor(s2, off, 32);
  }
  float mean = s * (1.f / 192.f);
  float var  = s2 * (1.f / 192.f) - mean * mean;
  float inv  = rsqrtf(var + 1e-5f);
  size_t obase;
  if (windowed) {
    int b = tok >> 12, rem = tok & 4095;
    int hh = rem >> 6, ww = rem & 63;
    int wt = ((b * 8 + (hh >> 3)) * 8 + (ww >> 3)) * 64 + (hh & 7) * 8 + (ww & 7);
    obase = (size_t)wt * 192;
  } else {
    obase = (size_t)tok * 192;
  }
#pragma unroll
  for (int j = 0; j < 6; ++j) {
    int c = lane + 32 * j;
    out[obase + c] = (_Float16)((vals[j] - mean) * inv * g[c] + bta[c]);
  }
}

// ---------------------------------------------------------------------------
// Tiled WMMA GEMM: Y[M,N] = A[M,K] (f16) @ W[K,N] (f16) + bias
//   EPI==0 : write f16 (KV projection)
//   EPI==1 : window-reverse + residual x, write f32 (attention proj -> y)
// Block 256 thr (8 waves), tile 128x64, k-step 32.  B staged transposed so
// each B fragment is two contiguous 16B LDS reads.
// ---------------------------------------------------------------------------
template <int EPI>
__global__ __launch_bounds__(256) void gemm_wmma(const _Float16* __restrict__ A,
                                                 const _Float16* __restrict__ W,
                                                 const float* __restrict__ bias,
                                                 const float* __restrict__ xres,
                                                 float* __restrict__ outf,
                                                 _Float16* __restrict__ outh,
                                                 int M, int N, int K) {
  __shared__ _Float16 As[128][32];   // [m][k]   16B-aligned rows
  __shared__ _Float16 BsT[64][32];   // [n][k]   transposed weight tile
  int t = threadIdx.x, lane = t & 31, wv = t >> 5;
  int m0 = blockIdx.y * 128, n0 = blockIdx.x * 64;
  v8f acc[4] = {};
  for (int k0 = 0; k0 < K; k0 += 32) {
    // stage A tile 128x32 (512 x 8-half chunks, 2 per thread), async when possible
#pragma unroll
    for (int i = 0; i < 2; ++i) {
      int lin = t + i * 256;
      int row = lin >> 2, ch = lin & 3;
      copy16_g2l(A + (size_t)(m0 + row) * K + k0 + ch * 8, &As[row][ch * 8]);
    }
    // stage B tile 32x64 transposed (1 x 8-half chunk per thread, scatter 8 b16)
    {
      int row = t >> 3, ch = t & 7;
      union { uint4 u; _Float16 h[8]; } uv;
      uv.u = *(const uint4*)(W + (size_t)(k0 + row) * N + n0 + ch * 8);
#pragma unroll
      for (int j = 0; j < 8; ++j) BsT[ch * 8 + j][row] = uv.h[j];
    }
    // prefetch next A tile into cache (global_prefetch_b8)
    if (k0 + 32 < K) {
      int row = t >> 2, ch = t & 3;
      __builtin_prefetch(A + (size_t)(m0 + row) * K + k0 + 32 + ch * 8, 0, 1);
    }
    wait_async_copies();
    __syncthreads();
    v16h a = frag_a(&As[wv * 16][0], 32, lane);
#pragma unroll
    for (int nt = 0; nt < 4; ++nt) {
      v16h b = frag_bT(&BsT[nt * 16][0], 32, lane);
      acc[nt] = WMMA_F16(a, b, acc[nt]);
    }
    __syncthreads();
  }
  int hi = lane >> 4, nlo = lane & 15;
#pragma unroll
  for (int nt = 0; nt < 4; ++nt) {
    int col = n0 + nt * 16 + nlo;
    float bv = bias[col];
#pragma unroll
    for (int e = 0; e < 8; ++e) {
      int row = m0 + wv * 16 + e + 8 * hi;
      float v = acc[nt][e] + bv;
      if (EPI == 0) {
        outh[(size_t)row * N + col] = (_Float16)v;
      } else {
        // window-reverse: windowed token -> spatial token, add residual x
        int w  = row >> 6, n = row & 63;
        int bi = w >> 6, wl = w & 63;
        int hh = (wl >> 3) * 8 + (n >> 3);
        int ww = (wl & 7) * 8 + (n & 7);
        size_t sp = ((size_t)(bi * 64 + hh) * 64 + ww) * 192 + col;
        outf[sp] = v + xres[sp];
      }
    }
  }
}

// ---------------------------------------------------------------------------
// Fused window attention: one block per (head, window); 128 thr = 4 waves.
// S = (q*scale) Kᵀ + relpos bias; softmax; O = P V.  V staged transposed.
// ---------------------------------------------------------------------------
__global__ __launch_bounds__(128) void attn_win(const float* __restrict__ qg,
                                                const _Float16* __restrict__ kv,
                                                const float* __restrict__ rpbt,
                                                _Float16* __restrict__ o16) {
  __shared__ _Float16 Qs[64][32];     // [n][d]
  __shared__ _Float16 Ks[64][32];     // [m][d]  (read transposed via frag_bT)
  __shared__ _Float16 VsT[32][64];    // [d][m]  (B-fragment ready)
  __shared__ float rpb[225];
  __shared__ _Float16 Ps[4][16][64];  // per-wave softmax probs
  int h = blockIdx.x;          // head 0..5
  int w = blockIdx.y;          // window 0..2047
  int t = threadIdx.x, lane = t & 31, wv = t >> 5;
  int bimg = w >> 6;           // 64 windows per image
  const float scale = 0.17677669529663687f;  // 1/sqrt(32)
  // K rows: contiguous 8-half chunks, async copy (256 chunks / 128 thr)
#pragma unroll
  for (int i = 0; i < 2; ++i) {
    int idx = t + i * 128;
    int n = idx >> 2, ch = idx & 3;
    copy16_g2l(kv + ((size_t)(w * 64 + n)) * 384 + h * 32 + ch * 8, &Ks[n][ch * 8]);
  }
  // Q (scaled f32->f16) and V (transposed) elementwise
  for (int idx = t; idx < 64 * 32; idx += 128) {
    int n = idx >> 5, d = idx & 31;
    Qs[n][d]  = (_Float16)(qg[(((size_t)bimg * 64 + n) * 6 + h) * 32 + d] * scale);
    VsT[d][n] = kv[((size_t)(w * 64 + n)) * 384 + 192 + h * 32 + d];
  }
  for (int idx = t; idx < 225; idx += 128) rpb[idx] = rpbt[idx * 6 + h];
  wait_async_copies();
  __syncthreads();

  int ms = wv * 16;
  v16h aq = frag_a(&Qs[ms][0], 32, lane);
  v8f st[4] = {};
#pragma unroll
  for (int nt = 0; nt < 4; ++nt) {
    v16h bk = frag_bT(&Ks[nt * 16][0], 32, lane);   // B = Kᵀ
    st[nt] = WMMA_F16(aq, bk, st[nt]);
  }
  int hi = lane >> 4, nlo = lane & 15;
  // bias + row softmax (rows live in 16-lane halves -> shfl_xor <= 8)
  float pr[4][8];
#pragma unroll
  for (int e = 0; e < 8; ++e) {
    int n = ms + e + 8 * hi;
    int i1 = n >> 3, j1 = n & 7;
    float mx = -1e30f;
#pragma unroll
    for (int nt = 0; nt < 4; ++nt) {
      int m = nt * 16 + nlo;
      int idx = (i1 - (m >> 3) + 7) * 15 + (j1 - (m & 7) + 7);
      float v = st[nt][e] + rpb[idx];
      pr[nt][e] = v;
      mx = fmaxf(mx, v);
    }
#pragma unroll
    for (int off = 8; off >= 1; off >>= 1) mx = fmaxf(mx, __shfl_xor(mx, off, 32));
    float sum = 0.f;
#pragma unroll
    for (int nt = 0; nt < 4; ++nt) {
      float p = __expf(pr[nt][e] - mx);
      pr[nt][e] = p; sum += p;
    }
#pragma unroll
    for (int off = 8; off >= 1; off >>= 1) sum += __shfl_xor(sum, off, 32);
    float rs = 1.f / sum;
    int r = e + 8 * hi;
#pragma unroll
    for (int nt = 0; nt < 4; ++nt)
      Ps[wv][r][nt * 16 + nlo] = (_Float16)(pr[nt][e] * rs);
  }
  // O = P @ V  (per-wave private P slot, same-wave dependency only)
  v8f oacc[2] = {};
#pragma unroll
  for (int kk = 0; kk < 2; ++kk) {
    v16h ap = frag_a(&Ps[wv][0][kk * 32], 64, lane);
#pragma unroll
    for (int dt = 0; dt < 2; ++dt) {
      v16h bv = frag_bT(&VsT[dt * 16][kk * 32], 64, lane);
      oacc[dt] = WMMA_F16(ap, bv, oacc[dt]);
    }
  }
#pragma unroll
  for (int dt = 0; dt < 2; ++dt)
#pragma unroll
    for (int e = 0; e < 8; ++e) {
      int n = ms + e + 8 * hi;
      o16[((size_t)(w * 64 + n)) * 192 + h * 32 + dt * 16 + nlo] = (_Float16)oacc[dt][e];
    }
}

// ---------------------------------------------------------------------------
// Fused MLP: out += fc2( gelu( fc1(h2) + b1 ) ) + b2, hidden staged in LDS
// in 32-wide chunks. Block 256 thr (8 waves), 64 rows per block.
// W1/W2 chunks staged transposed for contiguous B-fragment reads.
// ---------------------------------------------------------------------------
__global__ __launch_bounds__(256) void mlp_fused(const _Float16* __restrict__ h2,
                                                 const _Float16* __restrict__ w1,
                                                 const float* __restrict__ b1,
                                                 const _Float16* __restrict__ w2,
                                                 const float* __restrict__ b2,
                                                 float* out) {
  __shared__ _Float16 Ah[64][192];    // [m][k]
  __shared__ _Float16 W1T[32][192];   // [n][k]  fc1 weight chunk, transposed
  __shared__ _Float16 W2T[192][32];   // [n][k]  fc2 weight chunk, transposed
  __shared__ _Float16 Gs[64][32];     // gelu(fc1) chunk
  int t = threadIdx.x, lane = t & 31, wv = t >> 5;
  size_t m0 = (size_t)blockIdx.x * 64;
  // stage h2 tile 64x192 once (1536 x 8-half chunks, async when possible)
#pragma unroll
  for (int i = 0; i < 6; ++i) {
    int lin = t + i * 256;
    int row = lin / 24, ch = lin % 24;
    copy16_g2l(h2 + (m0 + row) * 192 + ch * 8, &Ah[row][ch * 8]);
  }
  wait_async_copies();
  int hi = lane >> 4, nlo = lane & 15;
  int mO = wv & 3, nH = wv >> 2;   // fc2 ownership: rows 16*mO, col half nH
  v8f oacc[6] = {};
  for (int chk = 0; chk < 24; ++chk) {
    __syncthreads();  // protect W1T/W2T/Gs from previous iteration readers
#pragma unroll
    for (int i = 0; i < 3; ++i) {  // W1 chunk 192x32, transpose into W1T
      int lin = t + i * 256;
      int row = lin >> 2, ch = lin & 3;    // row = k (0..191), ch*8 = n offset
      union { uint4 u; _Float16 h[8]; } uv;
      uv.u = *(const uint4*)(w1 + (size_t)row * 768 + chk * 32 + ch * 8);
#pragma unroll
      for (int j = 0; j < 8; ++j) W1T[ch * 8 + j][row] = uv.h[j];
    }
#pragma unroll
    for (int i = 0; i < 3; ++i) {  // W2 chunk 32x192, transpose into W2T
      int lin = t + i * 256;
      int row = lin / 24, ch = lin % 24;   // row = k (0..31), ch*8 = n offset
      union { uint4 u; _Float16 h[8]; } uv;
      uv.u = *(const uint4*)(w2 + (size_t)(chk * 32 + row) * 192 + ch * 8);
#pragma unroll
      for (int j = 0; j < 8; ++j) W2T[ch * 8 + j][row] = uv.h[j];
    }
    __syncthreads();
    // fc1: wave -> tile (mG, nG); G = gelu(A @ W1c + b1)
    int mG = wv >> 1, nG = wv & 1;
    v8f g = {};
#pragma unroll
    for (int kt = 0; kt < 6; ++kt) {
      v16h a = frag_a(&Ah[mG * 16][kt * 32], 192, lane);
      v16h b = frag_bT(&W1T[nG * 16][kt * 32], 192, lane);
      g = WMMA_F16(a, b, g);
    }
#pragma unroll
    for (int e = 0; e < 8; ++e) {
      int col = nG * 16 + nlo;
      float v = g[e] + b1[chk * 32 + col];
      v = 0.5f * v * (1.f + erff(v * 0.70710678118654752f));  // exact GELU
      Gs[mG * 16 + e + 8 * hi][col] = (_Float16)v;
    }
    __syncthreads();
    // fc2: out_acc += G_chunk @ W2_chunk (K=32, one WMMA per n-tile)
    v16h ag = frag_a(&Gs[mO * 16][0], 32, lane);
#pragma unroll
    for (int j = 0; j < 6; ++j) {
      v16h b = frag_bT(&W2T[(nH * 6 + j) * 16][0], 32, lane);
      oacc[j] = WMMA_F16(ag, b, oacc[j]);
    }
  }
  // epilogue: out = y + mlp + b2  (y already in out)
#pragma unroll
  for (int j = 0; j < 6; ++j) {
    int col = (nH * 6 + j) * 16 + nlo;
    float bv = b2[col];
#pragma unroll
    for (int e = 0; e < 8; ++e) {
      size_t row = m0 + mO * 16 + e + 8 * hi;
      size_t idx = row * 192 + col;
      out[idx] = out[idx] + oacc[j][e] + bv;
    }
  }
}

// ---------------------------------------------------------------------------
// Host launcher
// ---------------------------------------------------------------------------
extern "C" void kernel_launch(void* const* d_in, const int* in_sizes, int n_in,
                              void* d_out, int out_size, void* d_ws, size_t ws_size,
                              hipStream_t stream) {
  (void)in_sizes; (void)n_in; (void)out_size; (void)ws_size;
  const float* x     = (const float*)d_in[0];
  const float* qg    = (const float*)d_in[1];
  const float* n1g   = (const float*)d_in[2];
  const float* n1b   = (const float*)d_in[3];
  const float* qkvw  = (const float*)d_in[4];
  const float* qkvb  = (const float*)d_in[5];
  const float* rpbt  = (const float*)d_in[6];
  const float* projw = (const float*)d_in[7];
  const float* projb = (const float*)d_in[8];
  const float* n2g   = (const float*)d_in[9];
  const float* n2b   = (const float*)d_in[10];
  const float* fc1w  = (const float*)d_in[11];
  const float* fc1b  = (const float*)d_in[12];
  const float* fc2w  = (const float*)d_in[13];
  const float* fc2b  = (const float*)d_in[14];
  float* out = (float*)d_out;

  const int M = 32 * 64 * 64;  // 131072 tokens

  // workspace carve-out (all 256B aligned)
  char* p = (char*)d_ws;
  auto carve = [&](size_t bytes) {
    char* q = p;
    p += (bytes + 255) & ~(size_t)255;
    return q;
  };
  _Float16* wkv16  = (_Float16*)carve((size_t)192 * 384 * 2);
  _Float16* wprj16 = (_Float16*)carve((size_t)192 * 192 * 2);
  _Float16* wf1_16 = (_Float16*)carve((size_t)192 * 768 * 2);
  _Float16* wf2_16 = (_Float16*)carve((size_t)768 * 192 * 2);
  _Float16* h16    = (_Float16*)carve((size_t)M * 192 * 2);   // LN1 out, later LN2 out
  _Float16* kv16   = (_Float16*)carve((size_t)M * 384 * 2);   // K|V per windowed token
  _Float16* o16    = (_Float16*)carve((size_t)M * 192 * 2);   // attention output

  // 1) weights -> f16
  cvt_f16<<<(192 * 384 + 255) / 256, 256, 0, stream>>>(qkvw, wkv16, 192 * 384);
  cvt_f16<<<(192 * 192 + 255) / 256, 256, 0, stream>>>(projw, wprj16, 192 * 192);
  cvt_f16<<<(192 * 768 + 255) / 256, 256, 0, stream>>>(fc1w, wf1_16, 192 * 768);
  cvt_f16<<<(768 * 192 + 255) / 256, 256, 0, stream>>>(fc2w, wf2_16, 768 * 192);

  // 2) LN1 + window partition
  ln_kernel<<<M / 8, 256, 0, stream>>>(x, n1g, n1b, h16, 1);

  // 3) KV GEMM: [M,192] @ [192,384]
  gemm_wmma<0><<<dim3(384 / 64, M / 128), 256, 0, stream>>>(
      h16, wkv16, qkvb, nullptr, nullptr, kv16, M, 384, 192);

  // 4) fused window attention (2048 windows x 6 heads)
  attn_win<<<dim3(6, 2048), 128, 0, stream>>>(qg, kv16, rpbt, o16);

  // 5) proj GEMM + window reverse + residual -> y (f32 in d_out)
  gemm_wmma<1><<<dim3(192 / 64, M / 128), 256, 0, stream>>>(
      o16, wprj16, projb, x, out, nullptr, M, 192, 192);

  // 6) LN2 on y (spatial layout), reuse h16
  ln_kernel<<<M / 8, 256, 0, stream>>>(out, n2g, n2b, h16, 0);

  // 7) fused MLP, RMW on d_out: out = y + fc2(gelu(fc1(h2)))
  mlp_fused<<<M / 64, 256, 0, stream>>>(h16, wf1_16, fc1b, wf2_16, fc2b, out);
}
